// Attention3Context_6828998001316
// MI455X (gfx1250) — matrix operand order repaired and checked by reading, not compiled, not run on hardware
//
#include <hip/hip_runtime.h>
#include <stdint.h>

// ---------------------------------------------------------------------------
// MI455X (gfx1250) 3-context cross-attention, bf16 WMMA pipeline, round 3:
//  - weights pre-packed to bf16 pairs in WMMA B-fragment order
//  - activations pre-converted to bf16 (read once as f32)
//  - flash attention with TDM (tensor_load_to_lds) double-buffered K/V tiles
//  - row-sum via WMMA-with-ones, row-max via v_permlane16 butterflies
//  - V-projection transposed epilogue emits b128 stores (8 rows packed in-lane)
// ---------------------------------------------------------------------------

typedef __attribute__((ext_vector_type(16))) __bf16    v16bf;
typedef __attribute__((ext_vector_type(8)))  float     v8f;
typedef __attribute__((ext_vector_type(4)))  uint32_t  u32x4;
typedef __attribute__((ext_vector_type(4)))  int       i32x4;
typedef __attribute__((ext_vector_type(8)))  int       i32x8;

union BF16Frag {
  v16bf v;
  uint32_t u[8];
};

static constexpr int Bc     = 4;
static constexpr int NQc    = 1024;
static constexpr int NKTc   = 4096;   // 1024 + 1024 + 2048
static constexpr int INNERc = 512;    // H * DH
static constexpr int VDc    = 512;
static constexpr int DHc    = 64;
static constexpr float SCALEc = 0.125f;  // DH^-0.5

#ifndef USE_TDM
#if __has_builtin(__builtin_amdgcn_tensor_load_to_lds)
#define USE_TDM 1
#else
#define USE_TDM 0
#endif
#endif

__device__ __forceinline__ uint32_t pack2_bf16(float a, float b) {
  union { float f; uint32_t u; } ua, ub;
  ua.f = a; ub.f = b;
  uint32_t x = (ua.u + 0x7FFFu + ((ua.u >> 16) & 1u)) >> 16;   // RNE
  uint32_t y = (ub.u + 0x7FFFu + ((ub.u >> 16) & 1u)) >> 16;
  return (x & 0xFFFFu) | (y << 16);
}
__device__ __forceinline__ uint16_t f2bf(float a) {
  union { float f; uint32_t u; } ua; ua.f = a;
  return (uint16_t)((ua.u + 0x7FFFu + ((ua.u >> 16) & 1u)) >> 16);
}

// max-reduce + broadcast across each 16-lane group (wave32)
__device__ __forceinline__ float bcast_max16(float x) {
#if __has_builtin(__builtin_amdgcn_permlane16)
  union { float f; int i; } u, o;
  u.f = x;
  o.i = __builtin_amdgcn_permlane16(u.i, u.i, (int)0x67452301u, (int)0xefcdab89u, false, false); // xor 1
  u.f = fmaxf(u.f, o.f);
  o.i = __builtin_amdgcn_permlane16(u.i, u.i, (int)0x54761032u, (int)0xdcfe98bau, false, false); // xor 2
  u.f = fmaxf(u.f, o.f);
  o.i = __builtin_amdgcn_permlane16(u.i, u.i, (int)0x32107654u, (int)0xba98fedcu, false, false); // xor 4
  u.f = fmaxf(u.f, o.f);
  o.i = __builtin_amdgcn_permlane16(u.i, u.i, (int)0xfedcba98u, (int)0x76543210u, false, false); // xor 8
  return fmaxf(u.f, o.f);
#else
  #pragma unroll
  for (int off = 8; off >= 1; off >>= 1) x = fmaxf(x, __shfl_xor(x, off, 32));
  return x;
#endif
}

// ---------------------------------------------------------------------------
// One-time prep kernels
// ---------------------------------------------------------------------------
__global__ __launch_bounds__(256) void cvt_f32_bf16_kernel(
    const float* __restrict__ in, uint32_t* __restrict__ out, int npairs)
{
  const int i = blockIdx.x * blockDim.x + threadIdx.x;
  if (i < npairs) {
    const float2 f = ((const float2*)in)[i];
    out[i] = pack2_bf16(f.x, f.y);
  }
}

// weight [K][512] f32 -> [K/2][512] u32, word(k2,n) = pack(W[2k2][n], W[2k2+1][n])
__global__ __launch_bounds__(256) void wpack_kernel(
    const float* __restrict__ W, uint32_t* __restrict__ out, int nwords)
{
  const int i = blockIdx.x * blockDim.x + threadIdx.x;
  if (i < nwords) {
    const int n  = i & 511;
    const int k2 = i >> 9;
    out[i] = pack2_bf16(W[(size_t)(2 * k2) * 512 + n],
                        W[(size_t)(2 * k2 + 1) * 512 + n]);
  }
}

// 3 bool mask segments -> [B*NQ][NKT/32] u32 bitmask (bit = keep)
__global__ __launch_bounds__(256) void pack_masks_kernel(
    const uint8_t* __restrict__ m1, const uint8_t* __restrict__ m2,
    const uint8_t* __restrict__ m3, uint32_t* __restrict__ out)
{
  const int idx  = blockIdx.x * blockDim.x + threadIdx.x;   // B*NQ*(NKT/32)
  const int word = idx & (NKTc / 32 - 1);
  const int row  = idx >> 7;
  const int j = word * 32;
  const uint8_t* src; int segN, jl;
  if (j < 1024)      { src = m1; segN = 1024; jl = j;        }
  else if (j < 2048) { src = m2; segN = 1024; jl = j - 1024; }
  else               { src = m3; segN = 2048; jl = j - 2048; }
  const uint32_t* p = (const uint32_t*)(src + (size_t)row * segN + jl);
  uint32_t bits = 0;
  #pragma unroll
  for (int w = 0; w < 8; ++w) {
    const uint32_t u = p[w];
    bits |= ((u       ) & 1u) << (w * 4 + 0);
    bits |= ((u >>  8 ) & 1u) << (w * 4 + 1);
    bits |= ((u >> 16 ) & 1u) << (w * 4 + 2);
    bits |= ((u >> 24 ) & 1u) << (w * 4 + 3);
  }
  out[idx] = bits;
}

// ---------------------------------------------------------------------------
// WMMA GEMM: out = A(bf16) @ Wp(packed bf16) + bias. Wave: 32(M) x 64(N).
// ---------------------------------------------------------------------------
enum { OUT_BF16_ROW = 0, OUT_BF16_TRANS = 1, OUT_F32_ROW = 2 };

template<int OMODE>
__global__ __launch_bounds__(256) void gemm_wmma_kernel(
    const uint16_t* __restrict__ A, const uint32_t* __restrict__ Wp,
    const float* __restrict__ bias, void* __restrict__ outp,
    int Mtot, int K, int N, int M_pb,
    long long ob_stride, long long o_ld, long long o_off)
{
  const int lane = threadIdx.x & 31;
  const int wid  = threadIdx.x >> 5;
  const int lh   = lane >> 4;
  const int ln   = lane & 15;
  const int ncol = blockIdx.x * 64;
  const int rowbase = blockIdx.y * 256 + wid * 32;
  if (rowbase >= Mtot || ncol >= N) return;

  v8f c[2][4];
  #pragma unroll
  for (int mi = 0; mi < 2; ++mi)
    #pragma unroll
    for (int t = 0; t < 4; ++t)
      c[mi][t] = v8f{0.f,0.f,0.f,0.f,0.f,0.f,0.f,0.f};

  for (int k0 = 0; k0 < K; k0 += 32) {
    BF16Frag a[2];
    #pragma unroll
    for (int mi = 0; mi < 2; ++mi) {
      const uint32_t* rowA =
          (const uint32_t*)(A + (size_t)(rowbase + mi * 16 + ln) * K);
      #pragma unroll
      for (int v = 0; v < 8; ++v)
        a[mi].u[v] = rowA[(k0 >> 1) + lh * 8 + v];
    }
    BF16Frag bw[4];
    #pragma unroll
    for (int t = 0; t < 4; ++t) {
      const int col = ncol + t * 16 + ln;
      #pragma unroll
      for (int v = 0; v < 8; ++v)
        bw[t].u[v] = Wp[(size_t)((k0 >> 1) + lh * 8 + v) * N + col];
    }
    #pragma unroll
    for (int mi = 0; mi < 2; ++mi)
      #pragma unroll
      for (int t = 0; t < 4; ++t)
        c[mi][t] = __builtin_amdgcn_wmma_f32_16x16x32_bf16(
            false, a[mi].v, false, bw[t].v, (short)0, c[mi][t], false, false);
  }

  #pragma unroll
  for (int mi = 0; mi < 2; ++mi) {
    #pragma unroll
    for (int t = 0; t < 4; ++t) {
      const int col = ncol + t * 16 + ln;
      const float bv = bias ? bias[col] : 0.f;
      if constexpr (OMODE == OUT_BF16_TRANS) {
        // 8 consecutive output rows live in this lane -> one b128 store
        const int local0 = rowbase + mi * 16 + lh * 8;   // batch-local (M_pb | rows)
        const int batch  = local0 / M_pb;
        const int lloc   = local0 - batch * M_pb;
        uint32_t w[4];
        #pragma unroll
        for (int q = 0; q < 4; ++q)
          w[q] = pack2_bf16(c[mi][t][2 * q] + bv, c[mi][t][2 * q + 1] + bv);
        uint16_t* dst = (uint16_t*)outp +
            ((long long)batch * ob_stride + (long long)col * o_ld + o_off + lloc);
        *(uint4*)dst = *(const uint4*)w;
      } else {
        #pragma unroll
        for (int r = 0; r < 8; ++r) {
          const int row = rowbase + mi * 16 + lh * 8 + r;
          const int batch = row / M_pb;
          const int local = row - batch * M_pb;
          const float val = c[mi][t][r] + bv;
          if constexpr (OMODE == OUT_BF16_ROW) {
            ((uint16_t*)outp)[((long long)batch * ob_stride + o_off + local) * o_ld + col] = f2bf(val);
          } else {
            ((float*)outp)[((long long)batch * ob_stride + o_off + local) * o_ld + col] = val;
          }
        }
      }
    }
  }
}

// ---------------------------------------------------------------------------
// TDM 2D tile load: global (row-major, row stride in elements) -> LDS, bf16.
// D# encoding per CDNA5 ISA §8.3-8.5 (group0: count/lds/global addr/type,
// group1: data_size, tensor/tile dims, dim0 stride). Groups 2/3 zero (2D).
// ---------------------------------------------------------------------------
#if USE_TDM
__device__ __forceinline__ void tdm_load_2d_bf16(
    uint32_t lds_off, const uint16_t* gptr,
    uint32_t tile_d0, uint32_t tile_d1, uint64_t row_stride_elems)
{
  const uint64_t ga = (uint64_t)(uintptr_t)gptr;
  u32x4 g0;
  g0[0] = 1u;                                             // count=1, user desc
  g0[1] = lds_off;                                        // LDS byte address
  g0[2] = (uint32_t)ga;                                   // global_addr[31:0]
  g0[3] = (uint32_t)((ga >> 32) & 0x01FFFFFFu) | (2u << 30);  // addr[56:32]|type=2
  const uint32_t td1 = 1u << 20;                          // tensor_dim1 (OOB bound)
  i32x8 g1;
  g1[0] = (int)0x10000u;                                  // data_size=1 (2 bytes)
  g1[1] = (int)((tile_d0 & 0xFFFFu) << 16);               // tensor_dim0 = tile_d0
  g1[2] = (int)(((tile_d0 >> 16) & 0xFFFFu) | ((td1 & 0xFFFFu) << 16));
  g1[3] = (int)(((td1 >> 16) & 0xFFFFu) | ((tile_d0 & 0xFFFFu) << 16)); // tile_dim0
  g1[4] = (int)(tile_d1 & 0xFFFFu);                       // tile_dim1 (tile_dim2=0)
  g1[5] = (int)(uint32_t)(row_stride_elems & 0xFFFFFFFFu);// dim0 stride lo
  g1[6] = (int)(uint32_t)((row_stride_elems >> 32) & 0xFFFFu); // stride hi
  g1[7] = 0;
  i32x4 gz = {0, 0, 0, 0};
#if defined(__clang_major__) && __clang_major__ >= 23
  i32x8 gz8 = {0, 0, 0, 0, 0, 0, 0, 0};
  __builtin_amdgcn_tensor_load_to_lds(g0, g1, gz, gz, gz8, 0);
#else
  __builtin_amdgcn_tensor_load_to_lds(g0, g1, gz, gz, 0);
#endif
}
#endif

// ---------------------------------------------------------------------------
// Flash attention: block = (b,h) x 128 queries (8 waves x 16 q rows).
// 64-key chunks; K/V tiles staged in LDS (TDM double-buffered when available).
// ---------------------------------------------------------------------------
__global__ __launch_bounds__(256) void attn_fa_kernel(
    const uint16_t* __restrict__ Qbf, const uint16_t* __restrict__ Kbf,
    const uint16_t* __restrict__ Vt,  const uint32_t* __restrict__ mbits,
    uint16_t* __restrict__ attn_out)
{
#if USE_TDM
  __shared__ uint16_t kbuf[2][64 * 64];   // [key][dim]
  __shared__ uint16_t vbuf[2][64 * 64];   // [dim][key]
#else
  __shared__ uint16_t kbuf[1][64 * 64];
  __shared__ uint16_t vbuf[1][64 * 64];
#endif
  __shared__ uint16_t lds_p[8][16][64];   // per-wave P tiles

  const int lane = threadIdx.x & 31;
  const int wid  = threadIdx.x >> 5;
  const int lh   = lane >> 4;
  const int ln   = lane & 15;
  const int b = blockIdx.x >> 3;
  const int h = blockIdx.x & 7;
  const int qbase = blockIdx.y * 128 + wid * 16;

  // Q A-fragments (row M = ln), two K-steps covering DH=64
  const uint32_t* qrow = (const uint32_t*)(
      Qbf + ((size_t)(b * NQc) + qbase + ln) * INNERc + h * DHc);
  BF16Frag aq[2];
  #pragma unroll
  for (int ks = 0; ks < 2; ++ks)
    #pragma unroll
    for (int v = 0; v < 8; ++v)
      aq[ks].u[v] = qrow[ks * 16 + lh * 8 + v];

  BF16Frag ones;
  #pragma unroll
  for (int v = 0; v < 8; ++v) ones.u[v] = 0x3f803f80u;   // bf16 1.0 pairs

  v8f c[4];
  #pragma unroll
  for (int t = 0; t < 4; ++t) c[t] = v8f{0.f,0.f,0.f,0.f,0.f,0.f,0.f,0.f};
  float rmax[8], rsum[8];
  #pragma unroll
  for (int r = 0; r < 8; ++r) { rmax[r] = -3.0e38f; rsum[r] = 0.f; }

  const int mstride = NKTc / 32;
  const uint32_t* mrow_base =
      mbits + ((size_t)(b * NQc) + qbase + lh * 8) * mstride;

  const uint16_t* Kg = Kbf + ((size_t)(b * NKTc)) * INNERc + h * DHc; // stride INNERc
  const uint16_t* Vg = Vt + ((size_t)(b * INNERc) + h * DHc) * NKTc;  // stride NKTc

  const int nchunk = NKTc / 64;
#if USE_TDM
  if (wid == 0) {
    tdm_load_2d_bf16((uint32_t)(uintptr_t)&kbuf[0][0], Kg, 64, 64, INNERc);
    tdm_load_2d_bf16((uint32_t)(uintptr_t)&vbuf[0][0], Vg, 64, 64, NKTc);
  }
#endif

  for (int ci = 0; ci < nchunk; ++ci) {
    const int j = ci * 64;
#if USE_TDM
    const int cur = ci & 1;
    if (wid == 0) {
      if (ci + 1 < nchunk) {
        tdm_load_2d_bf16((uint32_t)(uintptr_t)&kbuf[cur ^ 1][0],
                         Kg + (size_t)(j + 64) * INNERc, 64, 64, INNERc);
        tdm_load_2d_bf16((uint32_t)(uintptr_t)&vbuf[cur ^ 1][0],
                         Vg + (j + 64), 64, 64, NKTc);
        __builtin_amdgcn_s_wait_tensorcnt(2);   // current tiles complete
      } else {
        __builtin_amdgcn_s_wait_tensorcnt(0);
      }
    }
    __syncthreads();
#else
    const int cur = 0;
    __syncthreads();
    for (int t = threadIdx.x; t < 64 * 8; t += 256) {  // 8 bf16 per slot
      const int row = t >> 3, seg = t & 7;
      *(uint4*)&kbuf[0][row * 64 + seg * 8] =
          *(const uint4*)(Kg + (size_t)(j + row) * INNERc + seg * 8);
      *(uint4*)&vbuf[0][row * 64 + seg * 8] =
          *(const uint4*)(Vg + (size_t)row * NKTc + j + seg * 8);
    }
    __syncthreads();
#endif

    // ---- S = Q K^T: 4 key tiles x 2 k-steps --------------------------------
    v8f s[4];
    #pragma unroll
    for (int t = 0; t < 4; ++t) {
      const uint32_t* krow = (const uint32_t*)&kbuf[cur][(t * 16 + ln) * 64];
      BF16Frag bk0, bk1;
      #pragma unroll
      for (int v = 0; v < 8; ++v) {
        bk0.u[v] = krow[lh * 8 + v];
        bk1.u[v] = krow[16 + lh * 8 + v];
      }
      v8f z = v8f{0.f,0.f,0.f,0.f,0.f,0.f,0.f,0.f};
      z = __builtin_amdgcn_wmma_f32_16x16x32_bf16(false, aq[0].v, false, bk0.v, (short)0, z, false, false);
      s[t] = __builtin_amdgcn_wmma_f32_16x16x32_bf16(false, aq[1].v, false, bk1.v, (short)0, z, false, false);
    }

    // ---- masked online softmax --------------------------------------------
    const uint32_t* mrow = mrow_base + (j >> 5);
    #pragma unroll
    for (int r = 0; r < 8; ++r) {
      const uint32_t mw0 = mrow[(size_t)r * mstride];
      const uint32_t mw1 = mrow[(size_t)r * mstride + 1];
      const bool km0 = (mw0 >> ln) & 1u;
      const bool km1 = (mw0 >> (16 + ln)) & 1u;
      const bool km2 = (mw1 >> ln) & 1u;
      const bool km3 = (mw1 >> (16 + ln)) & 1u;
      const float x0 = km0 ? s[0][r] * SCALEc : -1.0e30f;
      const float x1 = km1 ? s[1][r] * SCALEc : -1.0e30f;
      const float x2 = km2 ? s[2][r] * SCALEc : -1.0e30f;
      const float x3 = km3 ? s[3][r] * SCALEc : -1.0e30f;
      const float tmx = bcast_max16(fmaxf(fmaxf(x0, x1), fmaxf(x2, x3)));
      const float mold = rmax[r];
      const float mnew = fmaxf(mold, tmx);
      const float corr = __expf(mold - mnew);
      rmax[r] = mnew;
      rsum[r] *= corr;
      const float p0 = km0 ? __expf(x0 - mnew) : 0.f;
      const float p1 = km1 ? __expf(x1 - mnew) : 0.f;
      const float p2 = km2 ? __expf(x2 - mnew) : 0.f;
      const float p3 = km3 ? __expf(x3 - mnew) : 0.f;
      uint16_t* prow_st = &lds_p[wid][lh * 8 + r][0];
      prow_st[ln]      = f2bf(p0);
      prow_st[16 + ln] = f2bf(p1);
      prow_st[32 + ln] = f2bf(p2);
      prow_st[48 + ln] = f2bf(p3);
      #pragma unroll
      for (int t = 0; t < 4; ++t) c[t][r] *= corr;
    }
    asm volatile("s_wait_dscnt 0" ::: "memory");   // P stores -> A-frag loads

    // ---- P A-fragments + row sums via WMMA-with-ones -----------------------
    BF16Frag ap[2];
    const uint32_t* prow = (const uint32_t*)&lds_p[wid][ln][0];
    #pragma unroll
    for (int kc = 0; kc < 2; ++kc)
      #pragma unroll
      for (int v = 0; v < 8; ++v)
        ap[kc].u[v] = prow[kc * 16 + lh * 8 + v];
    v8f sum = v8f{0.f,0.f,0.f,0.f,0.f,0.f,0.f,0.f};
    sum = __builtin_amdgcn_wmma_f32_16x16x32_bf16(false, ap[0].v, false, ones.v, (short)0, sum, false, false);
    sum = __builtin_amdgcn_wmma_f32_16x16x32_bf16(false, ap[1].v, false, ones.v, (short)0, sum, false, false);
    #pragma unroll
    for (int r = 0; r < 8; ++r) rsum[r] += sum[r];

    // ---- O += P V ----------------------------------------------------------
    #pragma unroll
    for (int t = 0; t < 4; ++t) {
      const uint32_t* vrow = (const uint32_t*)&vbuf[cur][(t * 16 + ln) * 64];
      BF16Frag bv0, bv1;
      #pragma unroll
      for (int v = 0; v < 8; ++v) {
        bv0.u[v] = vrow[lh * 8 + v];
        bv1.u[v] = vrow[16 + lh * 8 + v];
      }
      c[t] = __builtin_amdgcn_wmma_f32_16x16x32_bf16(false, ap[0].v, false, bv0.v, (short)0, c[t], false, false);
      c[t] = __builtin_amdgcn_wmma_f32_16x16x32_bf16(false, ap[1].v, false, bv1.v, (short)0, c[t], false, false);
    }
#if USE_TDM
    __syncthreads();   // all waves done with buffer before wave0 overwrites it
#endif
  }

  // ---- normalize + store bf16 ----------------------------------------------
  #pragma unroll
  for (int r = 0; r < 8; ++r) {
    const float inv = rsum[r] > 0.f ? 1.f / rsum[r] : 0.f;
    const int row = qbase + lh * 8 + r;
    #pragma unroll
    for (int t = 0; t < 4; ++t)
      attn_out[((size_t)(b * NQc) + row) * VDc + h * DHc + t * 16 + ln] =
          f2bf(c[t][r] * inv);
  }
}

// ---------------------------------------------------------------------------
extern "C" void kernel_launch(void* const* d_in, const int* in_sizes, int n_in,
                              void* d_out, int out_size, void* d_ws, size_t ws_size,
                              hipStream_t stream) {
  (void)in_sizes; (void)n_in; (void)out_size; (void)ws_size;
  const float*   x    = (const float*)d_in[0];
  const float*   ctx1 = (const float*)d_in[1];
  const float*   ctx2 = (const float*)d_in[2];
  const float*   ctx3 = (const float*)d_in[3];
  const uint8_t* m1   = (const uint8_t*)d_in[4];
  const uint8_t* m2   = (const uint8_t*)d_in[5];
  const uint8_t* m3   = (const uint8_t*)d_in[6];
  const float* Wq  = (const float*)d_in[7];  const float* bq  = (const float*)d_in[8];
  const float* Wk1 = (const float*)d_in[9];  const float* bk1 = (const float*)d_in[10];
  const float* Wv1 = (const float*)d_in[11]; const float* bv1 = (const float*)d_in[12];
  const float* Wk2 = (const float*)d_in[13]; const float* bk2 = (const float*)d_in[14];
  const float* Wv2 = (const float*)d_in[15]; const float* bv2 = (const float*)d_in[16];
  const float* Wk3 = (const float*)d_in[17]; const float* bk3 = (const float*)d_in[18];
  const float* Wv3 = (const float*)d_in[19]; const float* bv3 = (const float*)d_in[20];
  const float* Wo  = (const float*)d_in[21]; const float* bo  = (const float*)d_in[22];
  float* out = (float*)d_out;

  char* ws = (char*)d_ws;
  uint16_t* Qbf  = (uint16_t*)ws; ws += (size_t)Bc * NQc  * INNERc * 2;   //  4 MB
  uint16_t* Kbf  = (uint16_t*)ws; ws += (size_t)Bc * NKTc * INNERc * 2;   // 16 MB
  uint16_t* Vt   = (uint16_t*)ws; ws += (size_t)Bc * INNERc * NKTc * 2;   // 16 MB
  uint32_t* mb   = (uint32_t*)ws; ws += (size_t)Bc * NQc * (NKTc/32) * 4; //  2 MB
  uint16_t* attn = (uint16_t*)ws; ws += (size_t)Bc * NQc * VDc * 2;       //  4 MB
  uint16_t* xbf  = (uint16_t*)ws; ws += (size_t)4096 * 512 * 2;
  uint16_t* c1bf = (uint16_t*)ws; ws += (size_t)4096 * 512 * 2;
  uint16_t* c2bf = (uint16_t*)ws; ws += (size_t)4096 * 768 * 2;
  uint16_t* c3bf = (uint16_t*)ws; ws += (size_t)8192 * 256 * 2;
  uint32_t* wq_p  = (uint32_t*)ws; ws += (size_t)256 * 512 * 4;
  uint32_t* wk1_p = (uint32_t*)ws; ws += (size_t)256 * 512 * 4;
  uint32_t* wk2_p = (uint32_t*)ws; ws += (size_t)384 * 512 * 4;
  uint32_t* wk3_p = (uint32_t*)ws; ws += (size_t)128 * 512 * 4;
  uint32_t* wv1_p = (uint32_t*)ws; ws += (size_t)256 * 512 * 4;
  uint32_t* wv2_p = (uint32_t*)ws; ws += (size_t)384 * 512 * 4;
  uint32_t* wv3_p = (uint32_t*)ws; ws += (size_t)128 * 512 * 4;
  uint32_t* wo_p  = (uint32_t*)ws; ws += (size_t)256 * 512 * 4;

  const dim3 blk(256);
  // activations -> bf16 (pairs)
  cvt_f32_bf16_kernel<<<dim3(4096), blk, 0, stream>>>(x,    (uint32_t*)xbf,  4096*512/2);
  cvt_f32_bf16_kernel<<<dim3(4096), blk, 0, stream>>>(ctx1, (uint32_t*)c1bf, 4096*512/2);
  cvt_f32_bf16_kernel<<<dim3(6144), blk, 0, stream>>>(ctx2, (uint32_t*)c2bf, 4096*768/2);
  cvt_f32_bf16_kernel<<<dim3(4096), blk, 0, stream>>>(ctx3, (uint32_t*)c3bf, 8192*256/2);
  // weights -> packed bf16 fragment order
  wpack_kernel<<<dim3(512), blk, 0, stream>>>(Wq,  wq_p,  256*512);
  wpack_kernel<<<dim3(512), blk, 0, stream>>>(Wk1, wk1_p, 256*512);
  wpack_kernel<<<dim3(768), blk, 0, stream>>>(Wk2, wk2_p, 384*512);
  wpack_kernel<<<dim3(256), blk, 0, stream>>>(Wk3, wk3_p, 128*512);
  wpack_kernel<<<dim3(512), blk, 0, stream>>>(Wv1, wv1_p, 256*512);
  wpack_kernel<<<dim3(768), blk, 0, stream>>>(Wv2, wv2_p, 384*512);
  wpack_kernel<<<dim3(256), blk, 0, stream>>>(Wv3, wv3_p, 128*512);
  wpack_kernel<<<dim3(512), blk, 0, stream>>>(Wo,  wo_p,  256*512);
  // mask bit-pack
  pack_masks_kernel<<<dim3((Bc * NQc * (NKTc / 32)) / 256), blk, 0, stream>>>(m1, m2, m3, mb);

  // Q / K projections -> bf16 row-major (K written into concat layout)
  gemm_wmma_kernel<OUT_BF16_ROW><<<dim3(8, 16), blk, 0, stream>>>(xbf,  wq_p,  bq,  Qbf, 4096, 512, 512, 1024, 1024, 512, 0);
  gemm_wmma_kernel<OUT_BF16_ROW><<<dim3(8, 16), blk, 0, stream>>>(c1bf, wk1_p, bk1, Kbf, 4096, 512, 512, 1024, 4096, 512, 0);
  gemm_wmma_kernel<OUT_BF16_ROW><<<dim3(8, 16), blk, 0, stream>>>(c2bf, wk2_p, bk2, Kbf, 4096, 768, 512, 1024, 4096, 512, 1024);
  gemm_wmma_kernel<OUT_BF16_ROW><<<dim3(8, 32), blk, 0, stream>>>(c3bf, wk3_p, bk3, Kbf, 8192, 256, 512, 2048, 4096, 512, 2048);
  // V projections -> bf16 transposed [B][512][NKT] (b128 epilogue stores)
  gemm_wmma_kernel<OUT_BF16_TRANS><<<dim3(8, 16), blk, 0, stream>>>(c1bf, wv1_p, bv1, Vt, 4096, 512, 512, 1024, (long long)INNERc * NKTc, 4096, 0);
  gemm_wmma_kernel<OUT_BF16_TRANS><<<dim3(8, 16), blk, 0, stream>>>(c2bf, wv2_p, bv2, Vt, 4096, 768, 512, 1024, (long long)INNERc * NKTc, 4096, 1024);
  gemm_wmma_kernel<OUT_BF16_TRANS><<<dim3(8, 32), blk, 0, stream>>>(c3bf, wv3_p, bv3, Vt, 8192, 256, 512, 2048, (long long)INNERc * NKTc, 4096, 2048);
  // flash attention
  attn_fa_kernel<<<dim3(Bc * 8, NQc / 128), blk, 0, stream>>>(Qbf, Kbf, Vt, mb, attn);
  // output projection -> f32
  gemm_wmma_kernel<OUT_F32_ROW><<<dim3(8, 16), blk, 0, stream>>>(attn, wo_p, bo, out, 4096, 512, 512, 4096, 4096, 512, 0);
}
